// CTFP2_64785286693516
// MI455X (gfx1250) — matrix operand order repaired and checked by reading, not compiled, not run on hardware
//
#include <hip/hip_runtime.h>
#include <hip/hip_bf16.h>

#define DIMX 32
#define HID 128
#define ROWS 128      // points per block (8 waves x 16 rows)
#define K1 64         // padded f-layer1 input dim (t,z,1,x0..31,pad)
#define KG 32         // padded g-layer1 input dim
#define NSTEPS 16
#define NLAYERS 2

typedef __attribute__((ext_vector_type(16))) __bf16 v16bf;
typedef __attribute__((ext_vector_type(8)))  float  v8f;
typedef __attribute__((ext_vector_type(2)))  __bf16 v2bf;

union FragU { v16bf v; uint4 q[2]; };

__device__ __forceinline__ unsigned short f2bf(float f) {
  unsigned u = __float_as_uint(f);
  u += 0x7FFFu + ((u >> 16) & 1u);          // round-to-nearest-even
  return (unsigned short)(u >> 16);
}

// Packed f32x2 -> bf16x2 convert: single v_cvt_pk_bf16_f32 (assembler-verified).
__device__ __forceinline__ unsigned cvt_pk_bf16(float a, float b) {
#if __has_builtin(__builtin_amdgcn_cvt_pk_bf16_f32)
  union { v2bf v; unsigned u; } cv;
  cv.v = __builtin_amdgcn_cvt_pk_bf16_f32(a, b);
  return cv.u;
#else
  unsigned r;
  asm("v_cvt_pk_bf16_f32 %0, %1, %2" : "=v"(r) : "v"(a), "v"(b));
  return r;
#endif
}

__device__ __forceinline__ void store_bf16_pair(unsigned short* p0, unsigned short* p1,
                                                float a, float b) {
  unsigned u = cvt_pk_bf16(a, b);
  *p0 = (unsigned short)u;          // ds_store_b16
  *p1 = (unsigned short)(u >> 16);  // ds_store_b16_d16_hi
}

__device__ __forceinline__ float tanh_fast(float x) {
#if __has_builtin(__builtin_amdgcn_tanhf)
  return __builtin_amdgcn_tanhf(x);          // native v_tanh_f32 (TRANS op)
#else
  float e = __expf(2.0f * x);
  return 1.0f - 2.0f * __builtin_amdgcn_rcpf(e + 1.0f);
#endif
}

// A fragment: 16x32 bf16, row-major LDS buffer. Lane l holds row (l&15);
// elements 0..7 -> K = kb..kb+7, elements 8..15 -> K = 16+kb..16+kb+7, kb = 8*(l>=16).
__device__ __forceinline__ v16bf load_frag_A(const unsigned short* buf, int row,
                                             int stride, int kc, int lane) {
  int kb = (lane >> 4) << 3;
  FragU f;
  f.q[0] = *(const uint4*)(buf + row * stride + kc + kb);
  f.q[1] = *(const uint4*)(buf + row * stride + kc + 16 + kb);
  return f.v;
}
// B fragment: 32x16 bf16 from W^T [n][k] row-major LDS. Lane l holds col n=(l&15);
// elements 0..7 -> K = kc+kb..+7, elements 8..15 -> K = kc+16+kb..+7, kb = 8*(l>=16).
__device__ __forceinline__ v16bf load_frag_B(const unsigned short* wt, int nbase,
                                             int stride, int kc, int lane) {
  const unsigned short* p = wt + (nbase + (lane & 15)) * stride + kc + ((lane >> 4) << 3);
  FragU f;
  f.q[0] = *(const uint4*)(p);
  f.q[1] = *(const uint4*)(p + 16);
  return f.v;
}

__device__ __forceinline__ v8f wmma_bf16(v16bf a, v16bf b, v8f c) {
  return __builtin_amdgcn_wmma_f32_16x16x32_bf16(false, a, false, b, (short)0, c, false, false);
}

// Dense hidden layer: [16 x K] @ [K x 128] (+ bias), tanh, bf16 store to Out.
// HASBIAS=false: bias is folded into a weight row against a constant-1 input
// column, so C starts at 0 (folds to the WMMA inline-0 SRC2 operand).
template<int KC, bool HASBIAS>
__device__ __forceinline__ void layer_tanh8(const unsigned short* Abuf, int astride,
                                            const unsigned short* Wt, int wstride,
                                            const float* bias,
                                            unsigned short* Out, int ostride,
                                            int rowbase, int lane) {
  const int lanelo = lane & 15;
  const int hi8 = (lane >> 4) << 3;
  v16bf a[KC];
#pragma unroll
  for (int kc = 0; kc < KC; ++kc)
    a[kc] = load_frag_A(Abuf, rowbase + lanelo, astride, kc * 32, lane);
  for (int nt = 0; nt < 8; ++nt) {
    int col = nt * 16 + lanelo;
    v8f c;
    if (HASBIAS) {
      float b = bias[col];
#pragma unroll
      for (int r = 0; r < 8; ++r) c[r] = b;
    } else {
#pragma unroll
      for (int r = 0; r < 8; ++r) c[r] = 0.f;
    }
#pragma unroll
    for (int kc = 0; kc < KC; ++kc)
      c = wmma_bf16(a[kc], load_frag_B(Wt, nt * 16, wstride, kc * 32, lane), c);
    unsigned short* outp = Out + (rowbase + hi8) * ostride + col;
#pragma unroll
    for (int r = 0; r < 8; r += 2)
      store_bf16_pair(outp + r * ostride, outp + (r + 1) * ostride,
                      tanh_fast(c[r]), tanh_fast(c[r + 1]));
  }
}

__global__ __launch_bounds__(256) void CTFP2_node_layer(
    const float* __restrict__ xin, const float* __restrict__ zin,
    float* __restrict__ xout, float* __restrict__ zout,
    const float* __restrict__ fW1, const float* __restrict__ fb1,
    const float* __restrict__ fW2, const float* __restrict__ fb2,
    const float* __restrict__ fW3, const float* __restrict__ fb3,
    const float* __restrict__ gW1, const float* __restrict__ gb1,
    const float* __restrict__ gW2, const float* __restrict__ gb2,
    const float* __restrict__ gW3, const float* __restrict__ gb3) {
  __shared__ __align__(16) unsigned short sW1T[HID * K1];    // f W1^T (+bias row k=2)
  __shared__ __align__(16) unsigned short sW2T[HID * HID];
  __shared__ __align__(16) unsigned short sW3T[DIMX * HID];
  __shared__ __align__(16) unsigned short sGW1T[HID * KG];   // g W1^T (+bias row k=2)
  __shared__ __align__(16) unsigned short sGW2T[HID * HID];
  __shared__ __align__(16) unsigned short sGW3T[16 * HID];   // g W3^T, N padded to 16
  __shared__ __align__(16) unsigned short sXf[ROWS * K1];    // [t, z, 1, x0..x31, 0...]
  __shared__ __align__(16) unsigned short sH[ROWS * HID];    // activations (reused)
  __shared__ float sFb2[HID], sFb3[DIMX];
  __shared__ float sGb2[HID], sGb3v;

  const int tid = threadIdx.x;
  const int lane = tid & 31;
  const int wave = tid >> 5;
  const int lanelo = lane & 15;
  const int hi8 = (lane >> 4) << 3;
  const int rowbase = wave * 16;
  const long gbase = (long)blockIdx.x * ROWS;

  // ---- stage transposed bf16 weights into LDS (once) ----
  // fW1 rows: 0=t, 1..32=x0..x31, 33=z.  LDS k order: 0=t, 1=z, 2=bias, 3..34=x.
  for (int i = tid; i < HID * K1; i += 256) {
    int n = i >> 6, k = i & 63;
    float v = 0.f;
    if (k == 0) v = fW1[0 * HID + n];
    else if (k == 1) v = fW1[33 * HID + n];
    else if (k == 2) v = fb1[n];
    else if (k < 35) v = fW1[(k - 2) * HID + n];
    sW1T[i] = f2bf(v);
  }
  for (int i = tid; i < HID * HID; i += 256) {
    int n = i >> 7, k = i & 127;
    sW2T[i] = f2bf(fW2[k * HID + n]);
  }
  for (int i = tid; i < DIMX * HID; i += 256) {
    int n = i >> 7, k = i & 127;
    sW3T[i] = f2bf(fW3[k * DIMX + n]);
  }
  for (int i = tid; i < HID * KG; i += 256) {       // gW1 rows: 0=t, 1=z; k=2 -> bias
    int n = i >> 5, k = i & 31;
    float v = 0.f;
    if (k == 0) v = gW1[n];
    else if (k == 1) v = gW1[HID + n];
    else if (k == 2) v = gb1[n];
    sGW1T[i] = f2bf(v);
  }
  for (int i = tid; i < HID * HID; i += 256) {
    int n = i >> 7, k = i & 127;
    sGW2T[i] = f2bf(gW2[k * HID + n]);
  }
  for (int i = tid; i < 16 * HID; i += 256) {
    int n = i >> 7, k = i & 127;
    sGW3T[i] = f2bf(n == 0 ? gW3[k] : 0.f);
  }
  for (int i = tid; i < HID; i += 256) { sFb2[i] = fb2[i]; sGb2[i] = gb2[i]; }
  if (tid < DIMX) sFb3[tid] = fb3[tid];
  if (tid == 0) sGb3v = gb3[0];
  for (int i = tid; i < ROWS * 29; i += 256) {      // zero-pad Xf cols 35..63
    int r = i / 29, c = 35 + i % 29;
    sXf[r * K1 + c] = 0;
  }

  // ---- load initial state: registers in D layout + bf16 mirror in Xf ----
  v8f sx[2], ax[2];   // x state / RK4 accumulator
  v8f zsv, zav;       // z state / accumulator (meaningful on lanelo==0 lanes)
#pragma unroll
  for (int r = 0; r < 8; ++r) { zsv[r] = 0.f; zav[r] = 0.f; }
#pragma unroll
  for (int nt = 0; nt < 2; ++nt)
#pragma unroll
    for (int r = 0; r < 8; ++r) {
      int m = rowbase + hi8 + r;
      float v = xin[(gbase + m) * DIMX + nt * 16 + lanelo];
      sx[nt][r] = v;
      ax[nt][r] = 0.f;
      sXf[m * K1 + 3 + nt * 16 + lanelo] = f2bf(v);
    }
  if (lanelo == 0) {
#pragma unroll
    for (int r = 0; r < 8; ++r) {
      int m = rowbase + hi8 + r;
      float z = zin[gbase + m];
      zsv[r] = z;
      sXf[m * K1 + 0] = f2bf(0.f);     // ODE time starts at 0
      sXf[m * K1 + 1] = f2bf(z);
      sXf[m * K1 + 2] = f2bf(1.0f);    // constant-1 column feeding the bias row
    }
  }
  __syncthreads();

  const float dt = 1.0f / NSTEPS;
  const float dt6 = dt / 6.0f;

  auto do_eval = [&](int ei, float tnext) {
    // ---- f path: 64 -> 128 -> 128 -> 32 ----
    layer_tanh8<2, false>(sXf, K1, sW1T, K1, nullptr, sH, HID, rowbase, lane);
    __syncthreads();
    layer_tanh8<4, true>(sH, HID, sW2T, HID, sFb2, sH, HID, rowbase, lane);
    __syncthreads();
    {   // f layer3 + RK4 update of x channels, writes next y into Xf (bf16)
      v16bf a[4];
#pragma unroll
      for (int kc = 0; kc < 4; ++kc)
        a[kc] = load_frag_A(sH, rowbase + lanelo, HID, kc * 32, lane);
#pragma unroll
      for (int nt = 0; nt < 2; ++nt) {
        int col = nt * 16 + lanelo;
        float b = sFb3[col];
        v8f c;
#pragma unroll
        for (int r = 0; r < 8; ++r) c[r] = b;
#pragma unroll
        for (int kc = 0; kc < 4; ++kc)
          c = wmma_bf16(a[kc], load_frag_B(sW3T, nt * 16, HID, kc * 32, lane), c);
        unsigned short* xfp = sXf + (rowbase + hi8) * K1 + 3 + col;
        float yn[8];
#pragma unroll
        for (int r = 0; r < 8; ++r) {
          float k = c[r];
          float acc = (ei == 0) ? k : ax[nt][r] + ((ei == 3) ? k : 2.0f * k);
          ax[nt][r] = acc;
          float ynew;
          if (ei == 3) { ynew = sx[nt][r] + dt6 * acc; sx[nt][r] = ynew; }
          else          ynew = sx[nt][r] + ((ei == 2) ? dt : 0.5f * dt) * k;
          yn[r] = ynew;
        }
#pragma unroll
        for (int r = 0; r < 8; r += 2)
          store_bf16_pair(xfp + r * K1, xfp + (r + 1) * K1, yn[r], yn[r + 1]);
      }
    }
    __syncthreads();
    // ---- g path: (t,z) -> 128 -> 128 -> 1 ----
    // g1 reads Xf chunk0 (cols 0..31): t, z, 1.0, x0..x28; only k=0,1,2 nonzero.
    layer_tanh8<1, false>(sXf, K1, sGW1T, KG, nullptr, sH, HID, rowbase, lane);
    __syncthreads();
    layer_tanh8<4, true>(sH, HID, sGW2T, HID, sGb2, sH, HID, rowbase, lane);
    __syncthreads();
    {   // g layer3 (N padded to 16; col 0 valid) + RK4 update of z, write t,z cols
      v16bf a[4];
#pragma unroll
      for (int kc = 0; kc < 4; ++kc)
        a[kc] = load_frag_A(sH, rowbase + lanelo, HID, kc * 32, lane);
      float b = (lanelo == 0) ? sGb3v : 0.f;
      v8f c;
#pragma unroll
      for (int r = 0; r < 8; ++r) c[r] = b;
#pragma unroll
      for (int kc = 0; kc < 4; ++kc)
        c = wmma_bf16(a[kc], load_frag_B(sGW3T, 0, HID, kc * 32, lane), c);
      if (lanelo == 0) {
#pragma unroll
        for (int r = 0; r < 8; ++r) {
          float k = c[r];
          float acc = (ei == 0) ? k : zav[r] + ((ei == 3) ? k : 2.0f * k);
          zav[r] = acc;
          float ynew;
          if (ei == 3) { ynew = zsv[r] + dt6 * acc; zsv[r] = ynew; }
          else          ynew = zsv[r] + ((ei == 2) ? dt : 0.5f * dt) * k;
          int m = rowbase + hi8 + r;
          store_bf16_pair(&sXf[m * K1 + 0], &sXf[m * K1 + 1], tnext, ynew);
        }
      }
    }
    __syncthreads();
  };

#pragma unroll 1
  for (int s = 0; s < NSTEPS; ++s) {
    float t0 = s * dt;
    do_eval(0, t0 + 0.5f * dt);
    do_eval(1, t0 + 0.5f * dt);
    do_eval(2, t0 + dt);
    do_eval(3, t0 + dt);      // tnext of last eval of last step is unused
  }

  // ---- write back state ----
#pragma unroll
  for (int nt = 0; nt < 2; ++nt)
#pragma unroll
    for (int r = 0; r < 8; ++r) {
      int m = rowbase + hi8 + r;
      xout[(gbase + m) * DIMX + nt * 16 + lanelo] = sx[nt][r];
    }
  if (lanelo == 0) {
#pragma unroll
    for (int r = 0; r < 8; ++r)
      zout[gbase + rowbase + hi8 + r] = zsv[r];
  }
}

extern "C" void kernel_launch(void* const* d_in, const int* in_sizes, int n_in,
                              void* d_out, int out_size, void* d_ws, size_t ws_size,
                              hipStream_t stream) {
  (void)n_in; (void)out_size; (void)ws_size;
  const float* w   = (const float*)d_in[0];
  const float* t   = (const float*)d_in[1];
  const float* fW1 = (const float*)d_in[2];
  const float* fb1 = (const float*)d_in[3];
  const float* fW2 = (const float*)d_in[4];
  const float* fb2 = (const float*)d_in[5];
  const float* fW3 = (const float*)d_in[6];
  const float* fb3 = (const float*)d_in[7];
  const float* gW1 = (const float*)d_in[8];
  const float* gb1 = (const float*)d_in[9];
  const float* gW2 = (const float*)d_in[10];
  const float* gb2 = (const float*)d_in[11];
  const float* gW3 = (const float*)d_in[12];
  const float* gb3 = (const float*)d_in[13];
  float* out = (float*)d_out;
  float* zws = (float*)d_ws;          // carries z channel between the 2 layer launches

  long N = (long)in_sizes[0] / DIMX;  // 524288 points
  int blocks = (int)(N / ROWS);
  const int FIN = DIMX + 2;           // 34

  for (int l = 0; l < NLAYERS; ++l) {
    const float* xi = (l == 0) ? w : out;
    const float* zi = (l == 0) ? t : zws;
    CTFP2_node_layer<<<blocks, 256, 0, stream>>>(
        xi, zi, out, zws,
        fW1 + (long)l * FIN * HID, fb1 + (long)l * HID,
        fW2 + (long)l * HID * HID, fb2 + (long)l * HID,
        fW3 + (long)l * HID * DIMX, fb3 + (long)l * DIMX,
        gW1 + (long)l * 2 * HID,   gb1 + (long)l * HID,
        gW2 + (long)l * HID * HID, gb2 + (long)l * HID,
        gW3 + (long)l * HID,       gb3 + (long)l);
  }
}
